// RNNModel_37383395345020
// MI455X (gfx1250) — compile-verified, matrix operand order checked by reading, not run on hardware
//
#include <hip/hip_runtime.h>
#include <stdint.h>

#define DEVI __device__ __forceinline__

typedef __attribute__((ext_vector_type(8)))  __bf16 v8bf;
typedef __attribute__((ext_vector_type(16))) __bf16 v16bf;
typedef __attribute__((ext_vector_type(8)))  float  v8f;

static constexpr int S_  = 128;
static constexpr int Bn  = 32;
static constexpr int D_  = 1024;
static constexpr int H_  = 1024;
static constexpr int V_  = 32000;
static constexpr int SB  = S_ * Bn;    // 4096 rows (S*B)
static constexpr int FH  = 4 * H_;     // 4096 gate columns

DEVI unsigned short f32_to_bf16_bits(float x) {
  unsigned u = __float_as_uint(x);
  unsigned r = u + 0x7FFFu + ((u >> 16) & 1u);   // round-to-nearest-even
  return (unsigned short)(r >> 16);
}

// bf16 A/B fragment per ISA 16-bit matrix layout:
// lane holds {k0+8*half .. +7} and {k0+16+8*half .. +7} of its row.
DEVI v16bf load_frag(const unsigned short* rowptr, int ka) {
  v8bf lo = *(const v8bf*)(rowptr + ka);
  v8bf hi = *(const v8bf*)(rowptr + ka + 16);
  return __builtin_shufflevector(lo, hi, 0, 1, 2, 3, 4, 5, 6, 7,
                                         8, 9, 10, 11, 12, 13, 14, 15);
}

DEVI v16bf frag_from(v8bf lo, v8bf hi) {
  return __builtin_shufflevector(lo, hi, 0, 1, 2, 3, 4, 5, 6, 7,
                                         8, 9, 10, 11, 12, 13, 14, 15);
}

// ---------------- elementwise / data-movement kernels ----------------

__global__ void k_f32_to_bf16(const float* __restrict__ in,
                              unsigned short* __restrict__ out, int n) {
  int i = blockIdx.x * blockDim.x + threadIdx.x;
  int stride = gridDim.x * blockDim.x;
  for (; i < n; i += stride) out[i] = f32_to_bf16_bits(in[i]);
}

__global__ void k_embed_gather(const int* __restrict__ tok,
                               const float* __restrict__ emb_w,
                               unsigned short* __restrict__ Xb) {
  int row = blockIdx.x;                       // 0..SB-1  (= s*B + b)
  int t = tok[row];
  const float* src = emb_w + (size_t)t * D_;
  unsigned short* dst = Xb + (size_t)row * D_;
  for (int e = threadIdx.x; e < D_; e += blockDim.x)
    dst[e] = f32_to_bf16_bits(src[e]);
}

__global__ void k_init_state(const float* __restrict__ hidden,
                             unsigned short* __restrict__ hb,
                             float* __restrict__ c) {
  int i = blockIdx.x * blockDim.x + threadIdx.x;
  if (i < Bn * H_) {
    float v = hidden[i];
    hb[i] = f32_to_bf16_bits(v);
    c[i] = v;
  }
}

__global__ void k_copy_f32(const float* __restrict__ in,
                           float* __restrict__ out, int n) {
  int i = blockIdx.x * blockDim.x + threadIdx.x;
  if (i < n) out[i] = in[i];
}

// ---------------- WMMA GEMM with async LDS-staged B ----------------
// C[M,N] = A[M,K] * W[N,K]^T (+bias).
// Block = 256 thr = 8 waves; block tile = 128 rows x 128 cols.
// Wave w owns m-tile (bm*128 + w*16) x all 8 n-tiles; the 32-wide k-slice of
// B (128 rows x 64B) is staged into LDS with global_load_async_to_lds_b128,
// double-buffered, shared by all 8 waves.

template <bool NTSTORE>
__global__ void k_wmma_gemm_lds(const unsigned short* __restrict__ A,
                                const unsigned short* __restrict__ W,
                                const float* __restrict__ bias,
                                float* __restrict__ C,
                                int N, int K, int nbn) {
  constexpr int NT_TILES = 8;
  constexpr int ROWPAD = 40;                    // 80B row stride: 16B-aligned,
                                                // bank-conflict-free fragments
  __shared__ unsigned short ldsB[2][128][ROWPAD];

  const int tid = threadIdx.x;
  const int lane = tid & 31;
  const int wave = tid >> 5;                    // m-tile within block
  const int bm = blockIdx.x / nbn;
  const int bn = blockIdx.x % nbn;
  const int m0 = bm * 128 + wave * 16;
  const int n0 = bn * 128;
  const int half = lane >> 4;
  const int r = lane & 15;

  // staging: 128 rows x 4 chunks of 16B = 512 chunks; 2 per thread
  const int c0 = tid * 2, c1 = tid * 2 + 1;
  const int c0row = c0 >> 2, c0q = c0 & 3;
  const int c1row = c1 >> 2, c1q = c1 & 3;

  auto stage = [&](int buf, int k0) {
    unsigned l0 = (unsigned)(uintptr_t)&ldsB[buf][c0row][c0q * 8];
    unsigned l1 = (unsigned)(uintptr_t)&ldsB[buf][c1row][c1q * 8];
    const unsigned short* g0 = W + (size_t)(n0 + c0row) * K + k0 + c0q * 8;
    const unsigned short* g1 = W + (size_t)(n0 + c1row) * K + k0 + c1q * 8;
    asm volatile("global_load_async_to_lds_b128 %0, %1, off"
                 :: "v"(l0), "v"(g0) : "memory");
    asm volatile("global_load_async_to_lds_b128 %0, %1, off"
                 :: "v"(l1), "v"(g1) : "memory");
  };

  const unsigned short* arow = A + (size_t)(m0 + r) * K;
  v8f acc[NT_TILES] = {};

  stage(0, 0);
  int buf = 0;
  for (int k0 = 0; k0 < K; k0 += 32) {
    const bool has_next = (k0 + 32) < K;
    if (has_next) {
      stage(buf ^ 1, k0 + 32);
      asm volatile("s_wait_asynccnt 0x2" ::: "memory");  // current slice done
    } else {
      asm volatile("s_wait_asynccnt 0x0" ::: "memory");
    }
    __syncthreads();

    const int ka = k0 + half * 8;
    v16bf af = load_frag(arow, ka);
    __builtin_prefetch(arow + ka + 32, 0, 1);            // global_prefetch_b8
#pragma unroll
    for (int t = 0; t < NT_TILES; ++t) {
      const unsigned short* lrow = &ldsB[buf][t * 16 + r][0];
      v8bf blo = *(const v8bf*)(lrow + half * 8);        // ds_load_b128
      v8bf bhi = *(const v8bf*)(lrow + 16 + half * 8);   // ds_load_b128
      acc[t] = __builtin_amdgcn_wmma_f32_16x16x32_bf16(
          false, af, false, frag_from(blo, bhi), (short)0, acc[t], false, false);
    }
    __syncthreads();   // keep `buf` readable until everyone is done with it
    buf ^= 1;
  }

#pragma unroll
  for (int t = 0; t < NT_TILES; ++t) {
    const int n = n0 + t * 16 + r;
    const float bv = bias ? bias[n] : 0.0f;
#pragma unroll
    for (int rr = 0; rr < 8; ++rr) {
      const int m = m0 + half * 8 + rr;     // C/D layout: row = vgpr + 8*half
      float v = acc[t][rr] + bv;
      float* dst = &C[(size_t)m * N + n];
      if (NTSTORE) __builtin_nontemporal_store(v, dst);  // don't evict dec_w
      else *dst = v;
    }
  }
}

// ---------------- Fused LSTM recurrent step ----------------
// gates = h_prev @ w_hh^T (WMMA) + gates_x[s]; then i,f,g,o activations,
// cell update, h output (bf16 for next step + hs, f32 for final state).

__global__ void k_lstm_step(const unsigned short* __restrict__ hprev,  // [B,H] bf16
                            const unsigned short* __restrict__ whh,    // [4H,H] bf16
                            const float* __restrict__ gates_x,         // [SB,4H]
                            float* __restrict__ c,                     // [B,H]
                            unsigned short* __restrict__ hnext,        // [B,H] bf16
                            float* __restrict__ hf,                    // [B,H] f32
                            unsigned short* __restrict__ hs,           // [SB,H] bf16
                            int s) {
  const int lane = threadIdx.x & 31;
  const int wid = blockIdx.x * (blockDim.x >> 5) + (threadIdx.x >> 5); // 0..127
  const int mt = wid & 1;        // batch tile (32 rows -> 2 tiles)
  const int jt = wid >> 1;       // hidden col tile (1024/16 = 64)
  const int m0 = mt << 4;
  const int j0 = jt << 4;
  const int half = lane >> 4;
  const int r = lane & 15;

  const unsigned short* arow = hprev + (size_t)(m0 + r) * H_;
  v8f acc[4] = {};

  for (int k0 = 0; k0 < H_; k0 += 32) {
    const int ka = k0 + half * 8;
    v16bf af = load_frag(arow, ka);
#pragma unroll
    for (int g = 0; g < 4; ++g) {
      const unsigned short* wrow = whh + (size_t)(g * H_ + j0 + r) * H_;
      v16bf bfr = load_frag(wrow, ka);
      acc[g] = __builtin_amdgcn_wmma_f32_16x16x32_bf16(
          false, af, false, bfr, (short)0, acc[g], false, false);
    }
  }

#pragma unroll
  for (int rr = 0; rr < 8; ++rr) {
    const int m = m0 + half * 8 + rr;
    const int j = j0 + r;
    const float* gxr = gates_x + (size_t)(s * Bn + m) * FH;
    float iv = acc[0][rr] + gxr[0 * H_ + j];
    float fv = acc[1][rr] + gxr[1 * H_ + j];
    float gv = acc[2][rr] + gxr[2 * H_ + j];
    float ov = acc[3][rr] + gxr[3 * H_ + j];
    iv = 1.0f / (1.0f + __expf(-iv));
    fv = 1.0f / (1.0f + __expf(-fv));
    gv = tanhf(gv);
    ov = 1.0f / (1.0f + __expf(-ov));
    const int idx = m * H_ + j;
    float cn = fv * c[idx] + iv * gv;
    float hv = ov * tanhf(cn);
    c[idx] = cn;
    unsigned short hb = f32_to_bf16_bits(hv);
    hnext[idx] = hb;
    hf[idx] = hv;
    hs[(size_t)(s * Bn + m) * H_ + j] = hb;
  }
}

// ---------------- host launcher ----------------

extern "C" void kernel_launch(void* const* d_in, const int* in_sizes, int n_in,
                              void* d_out, int out_size, void* d_ws, size_t ws_size,
                              hipStream_t stream) {
  (void)in_sizes; (void)n_in; (void)out_size; (void)ws_size;
  const int*   tok    = (const int*)  d_in[0];
  const float* hidden = (const float*)d_in[1];
  const float* emb_w  = (const float*)d_in[2];
  const float* w_ih   = (const float*)d_in[3];
  const float* w_hh   = (const float*)d_in[4];
  const float* dec_w  = (const float*)d_in[5];
  const float* dec_b  = (const float*)d_in[6];
  float* out = (float*)d_out;

  char* base = (char*)d_ws;
  size_t off = 0;
  auto alloc = [&](size_t bytes) -> char* {
    char* p = base + off;
    off += (bytes + 255) & ~(size_t)255;
    return p;
  };
  unsigned short* Xb    = (unsigned short*)alloc((size_t)SB * D_ * 2);   // emb bf16
  unsigned short* wih_b = (unsigned short*)alloc((size_t)FH * D_ * 2);
  unsigned short* whh_b = (unsigned short*)alloc((size_t)FH * H_ * 2);
  unsigned short* dcw_b = (unsigned short*)alloc((size_t)V_ * H_ * 2);
  float*          gx    = (float*)        alloc((size_t)SB * FH * 4);    // gates_x
  unsigned short* hs_b  = (unsigned short*)alloc((size_t)SB * H_ * 2);   // all h outputs
  unsigned short* hbuf0 = (unsigned short*)alloc((size_t)Bn * H_ * 2);
  unsigned short* hbuf1 = (unsigned short*)alloc((size_t)Bn * H_ * 2);
  float*          cbuf  = (float*)        alloc((size_t)Bn * H_ * 4);
  float*          hf    = (float*)        alloc((size_t)Bn * H_ * 4);

  // 1) convert weights to bf16, gather embeddings, init state
  k_f32_to_bf16<<<1024, 256, 0, stream>>>(w_ih, wih_b, FH * D_);
  k_f32_to_bf16<<<1024, 256, 0, stream>>>(w_hh, whh_b, FH * H_);
  k_f32_to_bf16<<<4096, 256, 0, stream>>>(dec_w, dcw_b, V_ * H_);
  k_embed_gather<<<SB, 256, 0, stream>>>(tok, emb_w, Xb);
  k_init_state<<<(Bn * H_) / 256, 256, 0, stream>>>(hidden, hbuf0, cbuf);

  // 2) gates_x = emb @ w_ih^T for all timesteps (regular stores: re-read later)
  {
    int N = FH, K = D_;
    int nbm = SB / 128, nbn = N / 128;         // 32 x 32
    k_wmma_gemm_lds<false><<<nbm * nbn, 256, 0, stream>>>(
        Xb, wih_b, nullptr, gx, N, K, nbn);
  }

  // 3) sequential LSTM scan, ping-pong h buffers
  for (int s = 0; s < S_; ++s) {
    const unsigned short* hp = (s & 1) ? hbuf1 : hbuf0;
    unsigned short*       hn = (s & 1) ? hbuf0 : hbuf1;
    k_lstm_step<<<32, 128, 0, stream>>>(hp, whh_b, gx, cbuf, hn, hf, hs_b, s);
  }

  // 4) decoded = hs @ dec_w^T + dec_b -> d_out (NT stores: keep dec_w in L2)
  {
    int N = V_, K = H_;
    int nbm = SB / 128, nbn = N / 128;         // 32 x 250
    k_wmma_gemm_lds<true><<<nbm * nbn, 256, 0, stream>>>(
        hs_b, dcw_b, dec_b, out, N, K, nbn);
  }

  // 5) h_final (f32) after the decoded block
  k_copy_f32<<<(Bn * H_) / 256, 256, 0, stream>>>(hf, out + (size_t)SB * V_, Bn * H_);
}
